// VectorQuantizer_13795434955014
// MI455X (gfx1250) — matrix operand order repaired and checked by reading, not compiled, not run on hardware
//
#include <hip/hip_runtime.h>

typedef __attribute__((ext_vector_type(16))) _Float16 v16h;
typedef __attribute__((ext_vector_type(8)))  _Float16 v8h;
typedef __attribute__((ext_vector_type(8)))  float    v8f;

#define NUM_CODES 1024
#define CDIM      256
#define HW        1024              // 32*32 spatial per batch
#define NVEC      32768             // 32 * HW
#define QSIZE     (32 * 256 * 32 * 32)
#define M_TILE    128               // rows per block (8 waves x 16 rows)
#define KT_TILES  64                // 1024 codes / 16

// ---------------------------------------------------------------------------
// Kernel 0: codebook prep.  embT[c][k] = f16(emb[k][c]);  hn[k] = 0.5*||e_k||^2
// ---------------------------------------------------------------------------
__global__ __launch_bounds__(256) void vq_prep(const float* __restrict__ emb,
                                               _Float16* __restrict__ embT,
                                               float* __restrict__ hn) {
    int k = blockIdx.x;           // code id, 1024 blocks
    int c = threadIdx.x;          // 256 threads = CDIM
    float v = emb[(size_t)k * CDIM + c];
    embT[(size_t)c * NUM_CODES + k] = (_Float16)v;
    __shared__ float red[256];
    red[c] = v * v;
    __syncthreads();
    for (int s = 128; s > 0; s >>= 1) {
        if (c < s) red[c] += red[c + s];
        __syncthreads();
    }
    if (c == 0) hn[k] = 0.5f * red[0];
}

// ---------------------------------------------------------------------------
// Kernel 1: fused GEMM (f16 WMMA) + running argmax of (x.e - 0.5||e||^2)
//   grid = 256 blocks of 256 threads (8 waves); block handles 128 rows.
//   B tiles double-buffered in LDS via GLOBAL_LOAD_ASYNC_TO_LDS_B128
//   (ASYNCcnt), overlapping the next tile's stage with this tile's WMMAs.
// ---------------------------------------------------------------------------
__global__ __launch_bounds__(256) void vq_argmin(const float* __restrict__ x,
                                                 const _Float16* __restrict__ embT,
                                                 const float* __restrict__ hn,
                                                 int* __restrict__ idx_out,
                                                 float* __restrict__ out) {
    // Double-buffered B tile: [buf][c in 0..255][16 codes] f16 = 2 x 8KB
    __shared__ __align__(16) _Float16 lds_b[2][CDIM * 16];

    const int tid  = threadIdx.x;
    const int wave = tid >> 5;
    const int lane = tid & 31;

    const int n0  = blockIdx.x * M_TILE;        // first row of this block
    const int b   = n0 / HW;                    // batch index
    const int hw0 = n0 % HW;                    // 128 consecutive n stay in one batch
    const float* xbase = x + (size_t)b * CDIM * HW + hw0;   // + c*HW + m

    // ---- Load all 8 A fragments (16x32 f16 each) straight into registers.
    // ISA A layout (16-bit 16x32): lanes 0-15 -> M=lane, K elems {0..7,16..23};
    // lanes 16-31 -> M=lane-16, K elems {8..15,24..31}.
    v16h afrag[8];
    {
        const int mloc = lane & 15;
        const int koff = (lane < 16) ? 0 : 8;
        const float* ab = xbase + (wave * 16 + mloc);
#pragma unroll
        for (int cc = 0; cc < 8; ++cc) {
            const float* p = ab + (size_t)(cc * 32 + koff) * HW;
            v16h a;
#pragma unroll
            for (int j = 0; j < 8; ++j) {
                a[j]     = (_Float16)p[(size_t)j * HW];          // K = koff+j
                a[8 + j] = (_Float16)p[(size_t)(16 + j) * HW];   // K = koff+16+j
            }
            afrag[cc] = a;
        }
    }

    // ---- Per-lane async staging slots: block stages 512 x 16B per tile;
    // each wave covers chunks e = wave*64 + i*32 + lane, i in {0,1}.
    // chunk e: c = e>>1, half = e&1 ->
    //   LDS  : buf + c*32 + half*16              (bytes)
    //   Glob : embT + c*2048 + kt*32 + half*16   (bytes)
    unsigned           lds_slot[2];
    unsigned long long gsrc_slot[2];
    {
#pragma unroll
        for (int i = 0; i < 2; ++i) {
            int e    = wave * 64 + i * 32 + lane;
            int c    = e >> 1;
            int half = e & 1;
            lds_slot[i]  = (unsigned)(c * 32 + half * 16);
            gsrc_slot[i] = (unsigned long long)(const char*)embT
                         + (unsigned long long)(unsigned)(c * 2048 + half * 16);
        }
    }
    const unsigned ldsbase0 = (unsigned)(size_t)&lds_b[0][0];  // LDS byte offset
    const unsigned ldsbase1 = (unsigned)(size_t)&lds_b[1][0];

#define STAGE_ASYNC(kt_, base_)                                                \
    {                                                                          \
        _Pragma("unroll")                                                      \
        for (int i = 0; i < 2; ++i) {                                          \
            unsigned           dst = (base_) + lds_slot[i];                    \
            unsigned long long src = gsrc_slot[i] + (unsigned)((kt_) * 32);    \
            asm volatile("global_load_async_to_lds_b128 %0, %1, off"           \
                         :: "v"(dst), "v"(src) : "memory");                    \
        }                                                                      \
    }

    float bestv[8];
    int   besti[8];
#pragma unroll
    for (int r = 0; r < 8; ++r) { bestv[r] = -3.0e38f; besti[r] = 0; }

    const int nloc = lane & 15;   // this lane's code column inside a 16-code tile

    STAGE_ASYNC(0, ldsbase0);     // prologue: stage tile 0 into buf 0

    for (int kt = 0; kt < KT_TILES; ++kt) {
        // my stage of tile kt landed (issued >=1 compute iteration ago)
        asm volatile("s_wait_asynccnt 0x0" ::: "memory");
        // all waves: stage(kt) landed AND reads of buf[(kt+1)&1] (iter kt-1) done
        __syncthreads();
        if (kt + 1 < KT_TILES) {
            STAGE_ASYNC(kt + 1, ((kt + 1) & 1) ? ldsbase1 : ldsbase0);
        }

        // 8 chained WMMAs: acc[m][n] = sum_c x[m][c] * emb[n][c]
        v8f acc = {};
        const _Float16* bb = &lds_b[kt & 1][((lane >> 4) * 8) * 16 + nloc];  // N = nloc, K base = 8*(lane/16)
#pragma unroll
        for (int cc = 0; cc < 8; ++cc) {
            v16h bfrag;
#pragma unroll
            for (int j = 0; j < 8; ++j) {
                bfrag[j]     = bb[cc * 512 + j * 16];
                bfrag[8 + j] = bb[cc * 512 + (16 + j) * 16];
            }
            acc = __builtin_amdgcn_wmma_f32_16x16x32_f16(
                false, afrag[cc], false, bfrag, (short)0, acc, false, false);
        }

        const int code = kt * 16 + nloc;
        const float h  = hn[code];
#pragma unroll
        for (int r = 0; r < 8; ++r) {
            float s = acc[r] - h;               // x.e - 0.5||e||^2
            if (s > bestv[r]) { bestv[r] = s; besti[r] = code; }
        }
    }
#undef STAGE_ASYNC

    // Cross-lane argmax over the 16 lanes that share each row (tie -> lower idx)
#pragma unroll
    for (int r = 0; r < 8; ++r) {
        float v = bestv[r];
        int   i = besti[r];
#pragma unroll
        for (int mask = 1; mask < 16; mask <<= 1) {
            float v2 = __shfl_xor(v, mask, 32);
            int   i2 = __shfl_xor(i, mask, 32);
            if (v2 > v || (v2 == v && i2 < i)) { v = v2; i = i2; }
        }
        bestv[r] = v; besti[r] = i;
    }

    // D layout: VGPR r holds M=r (lanes 0-15) / M=r+8 (lanes 16-31)
    if ((lane & 15) == 0) {
        const int mhalf = (lane >> 4) * 8;
#pragma unroll
        for (int r = 0; r < 8; ++r) {
            int n = n0 + wave * 16 + mhalf + r;
            idx_out[n] = besti[r];
            out[QSIZE + 1 + n] = (float)besti[r];   // indices output (flat b,h,w)
        }
    }
}

// ---------------------------------------------------------------------------
// Kernel 2: gather quantized rows (fp32 exact), write NCHW, per-block SSE
// ---------------------------------------------------------------------------
__global__ __launch_bounds__(256) void vq_gather(const float* __restrict__ x,
                                                 const float* __restrict__ emb,
                                                 const int* __restrict__ idx,
                                                 float* __restrict__ out,
                                                 float* __restrict__ partial) {
    __shared__ int   sidx[M_TILE];
    __shared__ float red[256];
    const int tid = threadIdx.x;
    const int n0  = blockIdx.x * M_TILE;
    if (tid < M_TILE) sidx[tid] = idx[n0 + tid];
    __syncthreads();

    const int b   = n0 / HW;
    const int hw0 = n0 % HW;
    const size_t base = (size_t)b * CDIM * HW + hw0;

    float acc = 0.f;
    for (int i = 0; i < 128; ++i) {
        int e = i * 256 + tid;           // 32768 = 256 c * 128 m
        int c = e >> 7;
        int m = e & 127;
        float q  = emb[(size_t)sidx[m] * CDIM + c];
        size_t o = base + (size_t)c * HW + m;
        float xv = x[o];
        out[o] = q;                      // quantized in NCHW layout
        float d = q - xv;
        acc += d * d;
    }
    red[tid] = acc;
    __syncthreads();
    for (int s = 128; s > 0; s >>= 1) {
        if (tid < s) red[tid] += red[tid + s];
        __syncthreads();
    }
    if (tid == 0) partial[blockIdx.x] = red[0];
}

// ---------------------------------------------------------------------------
// Kernel 3: final loss = (1 + BETA) * SSE / QSIZE
// ---------------------------------------------------------------------------
__global__ __launch_bounds__(256) void vq_loss(const float* __restrict__ partial,
                                               float* __restrict__ out) {
    __shared__ float red[256];
    const int tid = threadIdx.x;
    red[tid] = partial[tid];
    __syncthreads();
    for (int s = 128; s > 0; s >>= 1) {
        if (tid < s) red[tid] += red[tid + s];
        __syncthreads();
    }
    if (tid == 0) out[QSIZE] = 1.25f * red[0] / (float)QSIZE;
}

// ---------------------------------------------------------------------------
extern "C" void kernel_launch(void* const* d_in, const int* in_sizes, int n_in,
                              void* d_out, int out_size, void* d_ws, size_t ws_size,
                              hipStream_t stream) {
    (void)in_sizes; (void)n_in; (void)out_size; (void)ws_size;
    const float* x   = (const float*)d_in[0];   // [32,256,32,32] f32
    const float* emb = (const float*)d_in[1];   // [1024,256] f32
    float* out = (float*)d_out;                 // quantized | loss | indices

    char* ws = (char*)d_ws;
    _Float16* embT   = (_Float16*)ws;                    // 512 KB
    float*    hn     = (float*)(ws + 512 * 1024);        //   4 KB
    int*      idxbuf = (int*)(ws + 516 * 1024);          // 128 KB
    float*    part   = (float*)(ws + 644 * 1024);        //   1 KB

    vq_prep  <<<NUM_CODES,      256, 0, stream>>>(emb, embT, hn);
    vq_argmin<<<NVEC / M_TILE,  256, 0, stream>>>(x, embT, hn, idxbuf, out);
    vq_gather<<<NVEC / M_TILE,  256, 0, stream>>>(x, emb, idxbuf, out, part);
    vq_loss  <<<1,              256, 0, stream>>>(part, out);
}
